// GFNO_Burgers_Static_2D_51376398795182
// MI455X (gfx1250) — matrix operand order repaired and checked by reading, not compiled, not run on hardware
//
#include <hip/hip_runtime.h>
#include <hip/hip_fp16.h>
#include <math.h>

typedef _Float16 h16;
typedef __attribute__((ext_vector_type(8)))  _Float16 v8h;
typedef __attribute__((ext_vector_type(16))) _Float16 v16h;
typedef __attribute__((ext_vector_type(8)))  float    v8f;

#define PI_F 3.14159265358979323846f

// D = A(16x32 f16) * B(32x16 f16) + C(16x16 f32)
__device__ __forceinline__ v8f wmma_f16(v16h a, v16h b, v8f c) {
  return __builtin_amdgcn_wmma_f32_16x16x32_f16(false, a, false, b, (short)0, c,
                                                false, false);
}

__device__ __forceinline__ v16h pack16(v8h lo, v8h hi) {
  union { v16h v; v8h h[2]; } u;
  u.h[0] = lo; u.h[1] = hi;
  return u.v;
}

// A fragment: matrix stored [M][K] (row stride `stride` halves).
// lane<16: K = k0+{0..7, 16..23}; lane>=16: +8 (per CDNA5 16-bit A layout)
template <typename PT>
__device__ __forceinline__ v16h load_a16(PT base, int stride, int m0, int k0) {
  const int lane = threadIdx.x & 31;
  const int r = lane & 15, hs = lane >> 4;
  PT p = base + (size_t)(m0 + r) * stride + k0 + 8 * hs;
  v8h lo = *(const v8h*)(p);
  v8h hi = *(const v8h*)(p + 16);
  return pack16(lo, hi);
}

// B fragment: matrix stored transposed [N][K] (row stride `stride` halves).
// lane<16: K = k0+{0..7, 16..23} of column n0+lane; lane>=16: +8
template <typename PT>
__device__ __forceinline__ v16h load_b16(PT base, int stride, int n0, int k0) {
  const int lane = threadIdx.x & 31;
  const int r = lane & 15, hs = lane >> 4;
  PT p = base + (size_t)(n0 + r) * stride + k0 + 8 * hs;
  v8h lo = *(const v8h*)(p);
  v8h hi = *(const v8h*)(p + 16);
  return pack16(lo, hi);
}

__device__ __forceinline__ float gelu_f(float x) {
  return 0.5f * x * (1.0f + erff(x * 0.70710678118654752f));
}
__device__ __forceinline__ float sigm_f(float x) {
  return 1.0f / (1.0f + expf(-x));
}

// ---------------------------------------------------------------- tables
// ewR/ewI [ky16][w256]        : e^{-2pi i ky w/256}  (B for fwd stage1)
// ehA     [kx32][2h+ri, 512]  : e^{-2pi i f(kx) h/256} packed       (A stage2)
// fhA     [h256][2j+ri, 64]   : e^{+2pi i f(j) h/256} packed        (A zcalc)
// bwT     [w256][2ky+ri, 32]  : c_ky/(HW) * (cos, -sin)             (B inverse)
__global__ void k_tables(h16* ewR, h16* ewI, h16* ehA, h16* fhA, h16* bwT) {
  int t = blockIdx.x * 256 + threadIdx.x;
  const float TP = 2.0f * PI_F / 256.0f;
  if (t < 4096) {
    int ky = t >> 8, w = t & 255;
    int ph = (ky * w) & 255;
    ewR[t] = (h16)cosf(TP * (float)ph);
    ewI[t] = (h16)(-sinf(TP * (float)ph));
  }
  t -= 4096;
  if (t >= 0 && t < 16384) {
    int kx = t >> 9, rem = t & 511, h = rem >> 1, ri = rem & 1;
    int f = (kx < 16) ? kx : kx - 32;
    int ph = ((f * h) % 256 + 256) & 255;
    float th = TP * (float)ph;
    ehA[t] = (h16)(ri ? -sinf(th) : cosf(th));
  }
  t -= 16384;
  if (t >= 0 && t < 16384) {
    int h = t >> 6, rem = t & 63, j = rem >> 1, ri = rem & 1;
    int f = (j < 16) ? j : j - 32;
    int ph = ((f * h) % 256 + 256) & 255;
    float th = TP * (float)ph;
    fhA[t] = (h16)(ri ? sinf(th) : cosf(th));
  }
  t -= 16384;
  if (t >= 0 && t < 8192) {
    int w = t >> 5, rem = t & 31, ky = rem >> 1, ri = rem & 1;
    int ph = (ky * w) & 255;
    float th = TP * (float)ph;
    float s = ((ky == 0) ? 1.0f : 2.0f) / (256.0f * 256.0f);
    bwT[t] = (h16)(ri ? -s * sinf(th) : s * cosf(th));
  }
}

// hW [layer4][o32][i32] f16 conv weights (layer0 gets +I folded in)
// hFc1T [n128][c32] f16 transposed fc1 (B-format)
__global__ void k_wsmall(const float* w0, const float* w1, const float* w2,
                         const float* w3, const float* fc1w, h16* hW, h16* hFc1T) {
  int t = blockIdx.x * 256 + threadIdx.x;
  if (t < 4096) {
    int l = t >> 10, rem = t & 1023, o = rem >> 5, i = rem & 31;
    const float* w = (l == 0) ? w0 : (l == 1) ? w1 : (l == 2) ? w2 : w3;
    float v = w[o * 32 + i];
    if (l == 0 && o == i) v += 1.0f;
    hW[t] = (h16)v;
  } else if (t < 8192) {
    int rem = t - 4096, n = rem >> 5, c = rem & 31;
    hFc1T[rem] = (h16)fc1w[c * 128 + n];
  }
}

// spectral weights as B-format complex-packed tables:
// wR/wI [(l*32+j)*16+ky][o32][k=2i+ri, 64]
//   wR: (WR, -WI) rows ; wI: (WI, WR) rows
__global__ void k_wspec(const float* c1w1, const float* c1w2, const float* c2w1,
                        const float* c2w2, const float* c3w1, const float* c3w2,
                        h16* wR, h16* wI) {
  int t = blockIdx.x * 256 + threadIdx.x;
  const int TOT = 3 * 32 * 16 * 32 * 64;
  if (t >= TOT) return;
  int k = t & 63, o = (t >> 6) & 31, ky = (t >> 11) & 15, j = (t >> 15) & 31, l = t >> 20;
  int i = k >> 1, ri = k & 1;
  const float* w;
  if (l == 0)      w = (j < 16) ? c1w1 : c1w2;
  else if (l == 1) w = (j < 16) ? c2w1 : c2w2;
  else             w = (j < 16) ? c3w1 : c3w2;
  int kxp = (j < 16) ? j : j - 16;
  size_t si = ((((size_t)i * 32 + o) * 16 + kxp) * 16 + ky) * 2;
  float vR = w[si], vI = w[si + 1];
  wR[t] = (h16)(ri ? -vI : vR);
  wI[t] = (h16)(ri ? vR : vI);
}

// ---------------------------------------------------------------- CFT + gate
__global__ void k_cft(const float* x, const float* gw1, const float* gb1,
                      const float* gw2, const float* gb2, float* gate) {
  __shared__ float part[256 * 32];
  __shared__ float spec[32];
  __shared__ float g1[128];
  const int b = blockIdx.x, t = threadIdx.x;
  float zR[16], zI[16], dR[16], dI[16], aR[16], aI[16];
  const float p0 = -2.0f * PI_F * (float)t / 65535.0f;
  const float ps = -2.0f * PI_F * 256.0f / 65535.0f;
#pragma unroll
  for (int m = 0; m < 16; ++m) {
    float mm = (float)(m + 1);
    zR[m] = cosf(p0 * mm); zI[m] = sinf(p0 * mm);
    dR[m] = cosf(ps * mm); dI[m] = sinf(ps * mm);
    aR[m] = 0.f; aI[m] = 0.f;
  }
  const float* xb = x + (size_t)b * 65536;
#pragma unroll 1
  for (int j = t; j < 65536; j += 256) {
    float s = xb[j];
#pragma unroll
    for (int m = 0; m < 16; ++m) {
      aR[m] += s * zR[m]; aI[m] += s * zI[m];
      float nR = zR[m] * dR[m] - zI[m] * dI[m];
      float nI = zR[m] * dI[m] + zI[m] * dR[m];
      zR[m] = nR; zI[m] = nI;
    }
  }
#pragma unroll
  for (int m = 0; m < 16; ++m) {
    part[t * 32 + 2 * m] = aR[m];
    part[t * 32 + 2 * m + 1] = aI[m];
  }
  __syncthreads();
  if (t < 32) {
    float s = 0.f;
    for (int q = 0; q < 256; ++q) s += part[q * 32 + t];
    spec[t] = s / 65536.0f;
  }
  __syncthreads();
  if (t < 128) {
    float s = gb1[t];
#pragma unroll
    for (int i = 0; i < 32; ++i) s += spec[i] * gw1[i * 128 + t];
    g1[t] = gelu_f(s);
  }
  __syncthreads();
  if (t < 32) {
    float s = gb2[t];
    for (int j = 0; j < 128; ++j) s += g1[j] * gw2[j * 32 + t];
    gate[b * 32 + t] = sigm_f(s);
  }
}

// ---------------------------------------------------------------- fc0 + grid
__global__ void k_fc0(const float* x, const float* fw, const float* fb, h16* act) {
  int b = blockIdx.x >> 8, h = blockIdx.x & 255, w = threadIdx.x;
  float xv = x[(size_t)b * 65536 + h * 256 + w];
  float gx = (float)h / 255.0f, gy = (float)w / 255.0f;
#pragma unroll
  for (int c = 0; c < 32; ++c) {
    float v = xv * fw[c] + gx * fw[32 + c] + gy * fw[64 + c] + fb[c];
    act[(((size_t)(b * 32 + c)) * 256 + h) * 256 + w] = (h16)v;
  }
}

// ---------------------------------------------------------------- fwd DFT
// Xp[(kx*16+ky)*16+b][k=2c+ri, 64] complex-packed truncated spectrum
__global__ void k_fwd_dft(const h16* act, const h16* ewR, const h16* ewI,
                          const h16* ehA, h16* Xp) {
  __shared__ __align__(16) h16 sBR[16 * 512];
  __shared__ __align__(16) h16 sBI[16 * 512];
  const int bc = blockIdx.x;  // b*32+c
  const int b = bc >> 5, c = bc & 31;
  const int lane = threadIdx.x & 31, wave = threadIdx.x >> 5;
  const int r = lane & 15, hs = lane >> 4;
  const h16* img = act + (size_t)bc * 65536;
  // stage 1: Y[h,ky] = sum_w img * e^{-i..}; each wave does row-tiles
  // m0 and m0+128 in one K pass so every basis fragment is loaded once
  // and consumed immediately (no LICM hoist -> no VGPR spills).
  {
    const int m0 = wave * 16;
    v8f yR0 = {}, yI0 = {}, yR1 = {}, yI1 = {};
#pragma unroll 1
    for (int k0 = 0; k0 < 256; k0 += 32) {
      v16h a0 = load_a16(img, 256, m0, k0);
      v16h a1 = load_a16(img, 256, m0 + 128, k0);
      v16h bR = load_b16(ewR, 256, 0, k0);
      v16h bI = load_b16(ewI, 256, 0, k0);
      yR0 = wmma_f16(a0, bR, yR0);
      yI0 = wmma_f16(a0, bI, yI0);
      yR1 = wmma_f16(a1, bR, yR1);
      yI1 = wmma_f16(a1, bI, yI1);
    }
#pragma unroll
    for (int v = 0; v < 8; ++v) {
      int h0 = m0 + v + 8 * hs;
      int h1 = h0 + 128;
      int ky = r;
      sBR[ky * 512 + 2 * h0] = (h16)yR0[v];
      sBR[ky * 512 + 2 * h0 + 1] = (h16)(-yI0[v]);
      sBI[ky * 512 + 2 * h0] = (h16)yI0[v];
      sBI[ky * 512 + 2 * h0 + 1] = (h16)yR0[v];
      sBR[ky * 512 + 2 * h1] = (h16)yR1[v];
      sBR[ky * 512 + 2 * h1 + 1] = (h16)(-yI1[v]);
      sBI[ky * 512 + 2 * h1] = (h16)yI1[v];
      sBI[ky * 512 + 2 * h1 + 1] = (h16)yR1[v];
    }
  }
  __syncthreads();
  // stage 2: Xhat[kx,ky] = sum_h Eh * Y  (complex packed over K=512)
  if (wave < 2) {
    const int m0 = wave * 16;
    v8f xR = {}, xI = {};
#pragma unroll 2
    for (int k0 = 0; k0 < 512; k0 += 32) {
      v16h a = load_a16(ehA, 512, m0, k0);
      v16h bR = load_b16((const h16*)sBR, 512, 0, k0);
      v16h bI = load_b16((const h16*)sBI, 512, 0, k0);
      xR = wmma_f16(a, bR, xR);
      xI = wmma_f16(a, bI, xI);
    }
#pragma unroll
    for (int v = 0; v < 8; ++v) {
      int kx = m0 + v + 8 * hs, ky = r;
      size_t off = (((size_t)(kx * 16 + ky)) * 16 + b) * 64 + 2 * c;
      Xp[off] = (h16)xR[v];
      Xp[off + 1] = (h16)xI[v];
    }
  }
}

// ---------------------------------------------------------------- layer0 gate
// BR/BI [(b*32+c)*16+ky][k=2j+ri, 64]: Yhat in B-format for zcalc
__global__ void k_gate_mul(const h16* Xp, const float* gate, h16* BR, h16* BI) {
  int t = blockIdx.x * 256 + threadIdx.x;  // 262144
  int ky = t & 15, j = (t >> 4) & 31, c = (t >> 9) & 31, b = t >> 14;
  float yR = 0.f, yI = 0.f;
  if (j < 16) {
    size_t off = (((size_t)(j * 16 + ky)) * 16 + b) * 64 + 2 * c;
    float XR = (float)Xp[off], XI = (float)Xp[off + 1];
    float gR = gate[b * 32 + 2 * j] - 1.0f;  // (gate - 1): identity folded out
    float gI = gate[b * 32 + 2 * j + 1];
    yR = gR * XR - gI * XI;
    yI = gR * XI + gI * XR;
  }
  size_t row = (((size_t)(b * 32 + c)) * 16 + ky) * 64;
  BR[row + 2 * j] = (h16)yR;
  BR[row + 2 * j + 1] = (h16)(-yI);
  BI[row + 2 * j] = (h16)yI;
  BI[row + 2 * j + 1] = (h16)yR;
}

// ---------------------------------------------------------------- spectral mix
__global__ void k_spec_mul(const h16* Xp, const h16* wR, const h16* wI,
                           h16* BR, h16* BI) {
  const int wid = blockIdx.x * 8 + (threadIdx.x >> 5);  // 0..511 = j*16+ky
  const int j = wid >> 4, ky = wid & 15;
  const int lane = threadIdx.x & 31, r = lane & 15, hs = lane >> 4;
  const h16* ab = Xp + ((size_t)wid) * 16 * 64;
  const h16* wRb = wR + ((size_t)wid) * 32 * 64;
  const h16* wIb = wI + ((size_t)wid) * 32 * 64;
  v8f OR0 = {}, OI0 = {}, OR1 = {}, OI1 = {};
#pragma unroll 1
  for (int k0 = 0; k0 < 64; k0 += 32) {
    v16h a = load_a16(ab, 64, 0, k0);
    v16h bR0 = load_b16(wRb, 64, 0, k0);
    v16h bI0 = load_b16(wIb, 64, 0, k0);
    v16h bR1 = load_b16(wRb, 64, 16, k0);
    v16h bI1 = load_b16(wIb, 64, 16, k0);
    OR0 = wmma_f16(a, bR0, OR0);
    OI0 = wmma_f16(a, bI0, OI0);
    OR1 = wmma_f16(a, bR1, OR1);
    OI1 = wmma_f16(a, bI1, OI1);
  }
#pragma unroll
  for (int v = 0; v < 8; ++v) {
    int bb = v + 8 * hs;
    {
      size_t row = (((size_t)(bb * 32 + r)) * 16 + ky) * 64;
      BR[row + 2 * j] = (h16)OR0[v];
      BR[row + 2 * j + 1] = (h16)(-OI0[v]);
      BI[row + 2 * j] = (h16)OI0[v];
      BI[row + 2 * j + 1] = (h16)OR0[v];
    }
    {
      size_t row = (((size_t)(bb * 32 + 16 + r)) * 16 + ky) * 64;
      BR[row + 2 * j] = (h16)OR1[v];
      BR[row + 2 * j + 1] = (h16)(-OI1[v]);
      BI[row + 2 * j] = (h16)OI1[v];
      BI[row + 2 * j + 1] = (h16)OR1[v];
    }
  }
}

// ---------------------------------------------------------------- Z = Fh @ Yhat
// Zp [b*32+o][h256][2ky+ri, 32]  (A-format for the fused inverse kernel)
__global__ void k_zcalc(const h16* BR, const h16* BI, const h16* fhA, h16* Zp) {
  const int wid = blockIdx.x * 8 + (threadIdx.x >> 5);  // 0..511 = b*32+o
  const int lane = threadIdx.x & 31, r = lane & 15, hs = lane >> 4;
  const h16* bRb = BR + (size_t)wid * 16 * 64;
  const h16* bIb = BI + (size_t)wid * 16 * 64;
  v16h bR0 = load_b16(bRb, 64, 0, 0);
  v16h bR1 = load_b16(bRb, 64, 0, 32);
  v16h bI0 = load_b16(bIb, 64, 0, 0);
  v16h bI1 = load_b16(bIb, 64, 0, 32);
  h16* zb = Zp + (size_t)wid * 256 * 32;
#pragma unroll 1
  for (int tile = 0; tile < 16; ++tile) {
    int m0 = tile * 16;
    v8f ZR = {}, ZI = {};
    v16h a0 = load_a16(fhA, 64, m0, 0);
    v16h a1 = load_a16(fhA, 64, m0, 32);
    ZR = wmma_f16(a0, bR0, ZR);
    ZR = wmma_f16(a1, bR1, ZR);
    ZI = wmma_f16(a0, bI0, ZI);
    ZI = wmma_f16(a1, bI1, ZI);
#pragma unroll
    for (int v = 0; v < 8; ++v) {
      int hh = m0 + v + 8 * hs, ky = r;
      zb[(size_t)hh * 32 + 2 * ky] = (h16)ZR[v];
      zb[(size_t)hh * 32 + 2 * ky + 1] = (h16)ZI[v];
    }
  }
}

// ------------------------------------------- fused inverse + conv1x1 + GELU
__global__ void k_inv_conv(const h16* Zp, const h16* bwT, const h16* hW,
                           const float* bias, const h16* actIn, h16* actOut,
                           int apply_gelu) {
  const int b = blockIdx.x >> 8, h = blockIdx.x & 255;
  const int lane = threadIdx.x & 31, wave = threadIdx.x >> 5;
  const int r = lane & 15, hs = lane >> 4;
  const h16* zb = Zp + ((size_t)b * 32) * 8192 + (size_t)h * 32;      // o-stride 8192
  const h16* inb = actIn + ((size_t)b * 32) * 65536 + (size_t)h * 256; // c-stride 65536
  h16* outb = actOut + ((size_t)b * 32) * 65536 + (size_t)h * 256;
#pragma unroll 1
  for (int q = 0; q < 2; ++q) {
    const int w0 = (wave * 2 + q) * 16;
    v16h bc;  // conv B-tile: in[c=K][h][w0+N]
#pragma unroll
    for (int jj = 0; jj < 16; ++jj) {
      int cc = (jj < 8) ? 8 * hs + jj : 16 + 8 * hs + (jj - 8);
      bc[jj] = inb[(size_t)cc * 65536 + w0 + r];
    }
    v16h bw = load_b16(bwT, 32, w0, 0);
#pragma unroll
    for (int mt = 0; mt < 2; ++mt) {
      const int m0 = mt * 16;
      v8f acc = {};
      v16h ai = load_a16(zb, 8192, m0, 0);  // Z rows for o = m0..m0+15
      v16h ac = load_a16(hW, 32, m0, 0);    // conv weights
      acc = wmma_f16(ai, bw, acc);          // inverse transform along w
      acc = wmma_f16(ac, bc, acc);          // + 1x1 conv (layer0: W+I)
#pragma unroll
      for (int v = 0; v < 8; ++v) {
        int o = m0 + v + 8 * hs;
        float val = acc[v] + bias[o];
        if (apply_gelu) val = gelu_f(val);
        outb[(size_t)o * 65536 + w0 + r] = (h16)val;
      }
    }
  }
}

// ---------------------------------------------------------------- head fc1/fc2
__global__ void k_head(const h16* act, const h16* fc1T, const float* fc1b,
                       const float* fc2w, const float* fc2b, float* out) {
  const int b = blockIdx.x >> 9;
  const int p0 = ((blockIdx.x & 511) * 8 + (threadIdx.x >> 5)) * 16;
  const int lane = threadIdx.x & 31, r = lane & 15, hs = lane >> 4;
  const h16* ab = act + ((size_t)b * 32) * 65536;
  v16h a;  // A-tile: 16 pixels x 32 channels
#pragma unroll
  for (int jj = 0; jj < 8; ++jj) {
    int c0 = jj + 8 * hs;
    a[jj] = ab[(size_t)c0 * 65536 + p0 + r];
    a[jj + 8] = ab[(size_t)(c0 + 16) * 65536 + p0 + r];
  }
  float partial[8];
#pragma unroll
  for (int v = 0; v < 8; ++v) partial[v] = 0.f;
#pragma unroll 2
  for (int nt = 0; nt < 8; ++nt) {
    int n0 = nt * 16;
    v16h bb = load_b16(fc1T, 32, n0, 0);
    v8f acc = {};
    acc = wmma_f16(a, bb, acc);
    int n = n0 + r;
    float w2 = fc2w[n], bs = fc1b[n];
#pragma unroll
    for (int v = 0; v < 8; ++v) partial[v] += gelu_f(acc[v] + bs) * w2;
  }
#pragma unroll
  for (int v = 0; v < 8; ++v) {
    float s = partial[v];
    s += __shfl_xor(s, 1, 16);
    s += __shfl_xor(s, 2, 16);
    s += __shfl_xor(s, 4, 16);
    s += __shfl_xor(s, 8, 16);
    if (r == 0) {
      int p = p0 + v + 8 * hs;
      out[(size_t)b * 65536 + p] = s + fc2b[0];
    }
  }
}

// ----------------------------------------------------------------------------
extern "C" void kernel_launch(void* const* d_in, const int* in_sizes, int n_in,
                              void* d_out, int out_size, void* d_ws, size_t ws_size,
                              hipStream_t stream) {
  (void)in_sizes; (void)n_in; (void)out_size; (void)ws_size;
  const float* x    = (const float*)d_in[0];
  const float* fc0w = (const float*)d_in[1];
  const float* fc0b = (const float*)d_in[2];
  const float* gw1  = (const float*)d_in[3];
  const float* gb1  = (const float*)d_in[4];
  const float* gw2  = (const float*)d_in[5];
  const float* gb2  = (const float*)d_in[6];
  const float* c1w1 = (const float*)d_in[7];
  const float* c1w2 = (const float*)d_in[8];
  const float* c2w1 = (const float*)d_in[9];
  const float* c2w2 = (const float*)d_in[10];
  const float* c3w1 = (const float*)d_in[11];
  const float* c3w2 = (const float*)d_in[12];
  const float* w0w = (const float*)d_in[13]; const float* w0b = (const float*)d_in[14];
  const float* w1w = (const float*)d_in[15]; const float* w1b = (const float*)d_in[16];
  const float* w2w = (const float*)d_in[17]; const float* w2b = (const float*)d_in[18];
  const float* w3w = (const float*)d_in[19]; const float* w3b = (const float*)d_in[20];
  const float* fc1w = (const float*)d_in[21]; const float* fc1b = (const float*)d_in[22];
  const float* fc2w = (const float*)d_in[23]; const float* fc2b = (const float*)d_in[24];
  float* out = (float*)d_out;

  h16* w = (h16*)d_ws;
  size_t off = 0;
  const size_t SACT = (size_t)16 * 32 * 65536;
  h16* actA = w + off; off += SACT;
  h16* actB = w + off; off += SACT;
  h16* ewR  = w + off; off += 4096;
  h16* ewI  = w + off; off += 4096;
  h16* ehA  = w + off; off += 16384;
  h16* fhA  = w + off; off += 16384;
  h16* bwT  = w + off; off += 8192;
  h16* Xp   = w + off; off += 524288;
  h16* BRb  = w + off; off += 524288;
  h16* BIb  = w + off; off += 524288;
  h16* Zp   = w + off; off += (size_t)16 * 32 * 256 * 32;
  h16* hW   = w + off; off += 4096;
  h16* fc1T = w + off; off += 4096;
  h16* wspR = w + off; off += (size_t)3 * 32 * 16 * 32 * 64;
  h16* wspI = w + off; off += (size_t)3 * 32 * 16 * 32 * 64;
  float* gate = (float*)(w + off);

  k_tables<<<dim3(256), dim3(256), 0, stream>>>(ewR, ewI, ehA, fhA, bwT);
  k_wsmall<<<dim3(32), dim3(256), 0, stream>>>(w0w, w1w, w2w, w3w, fc1w, hW, fc1T);
  k_wspec<<<dim3(12288), dim3(256), 0, stream>>>(c1w1, c1w2, c2w1, c2w2, c3w1, c3w2,
                                                 wspR, wspI);
  k_cft<<<dim3(16), dim3(256), 0, stream>>>(x, gw1, gb1, gw2, gb2, gate);
  k_fc0<<<dim3(4096), dim3(256), 0, stream>>>(x, fc0w, fc0b, actA);

  const h16* ins[4]      = {actA, actB, actA, actB};
  h16* outs[4]           = {actB, actA, actB, actA};
  const float* biases[4] = {w0b, w1b, w2b, w3b};
  for (int layer = 0; layer < 4; ++layer) {
    k_fwd_dft<<<dim3(512), dim3(256), 0, stream>>>(ins[layer], ewR, ewI, ehA, Xp);
    if (layer == 0) {
      k_gate_mul<<<dim3(1024), dim3(256), 0, stream>>>(Xp, gate, BRb, BIb);
    } else {
      k_spec_mul<<<dim3(64), dim3(256), 0, stream>>>(
          Xp, wspR + (size_t)(layer - 1) * 1048576,
          wspI + (size_t)(layer - 1) * 1048576, BRb, BIb);
    }
    k_zcalc<<<dim3(64), dim3(256), 0, stream>>>(BRb, BIb, fhA, Zp);
    k_inv_conv<<<dim3(4096), dim3(256), 0, stream>>>(
        Zp, bwT, hW + layer * 1024, biases[layer], ins[layer], outs[layer],
        (layer < 3) ? 1 : 0);
  }
  k_head<<<dim3(8192), dim3(256), 0, stream>>>(actA, fc1T, fc1b, fc2w, fc2b, out);
}
